// AegisHTGNN_80711025427257
// MI455X (gfx1250) — matrix operand order repaired and checked by reading, not compiled, not run on hardware
//
#include <hip/hip_runtime.h>
#include <hip/hip_bf16.h>

// ---------------------------------------------------------------------------
// AegisHTGNN forward for gfx1250 (MI455X). wave32, WMMA f32_16x16x32_f16.
// ---------------------------------------------------------------------------

typedef __attribute__((ext_vector_type(16))) _Float16 v16h;
typedef __attribute__((ext_vector_type(8)))  float    v8f;

#define NPAYER 200000
#define NMERCH 50000
#define NEDGE  1000000
#define DM 64

static __device__ __forceinline__ float sigmoidf_(float x) { return 1.f / (1.f + __expf(-x)); }
static __device__ __forceinline__ float geluf_(float x)    { return 0.5f * x * (1.f + erff(x * 0.7071067811865475f)); }

static __device__ __forceinline__ void atomicMaxF(float* a, float v) {
  if (v >= 0.f) atomicMax((int*)a, __float_as_int(v));
  else          atomicMin((unsigned int*)a, __float_as_uint(v));
}
static __device__ __forceinline__ void atomicMinF(float* a, float v) {
  if (v >= 0.f) atomicMin((int*)a, __float_as_int(v));
  else          atomicMax((unsigned int*)a, __float_as_uint(v));
}

// ---------------------------------------------------------------------------
// Weight packing into WMMA B-fragment layout (16-bit B 32x16 per k-tile):
// lane 0..15 -> N=lane, K = kt*32 + {0..7,16..23} ; lane 16..31 -> N=lane-16, K = kt*32+{8..15,24..31}
// packed index = ((ct*KT + kt)*32 + lane)*16 + h
// ---------------------------------------------------------------------------
__global__ void pack_b_kernel(const float* __restrict__ W, _Float16* __restrict__ out,
                              int Kvalid, int KT, int CT, int Ncols) {
  int t = blockIdx.x * blockDim.x + threadIdx.x;
  int total = CT * KT * 32 * 16;
  if (t >= total) return;
  int h = t & 15, lane = (t >> 4) & 31;
  int g = t >> 9;
  int kt = g % KT, ct = g / KT;
  int k   = kt * 32 + ((lane >> 4) << 3) + (h & 7) + ((h >> 3) << 4);
  int col = ct * 16 + (lane & 15);
  float v = (k < Kvalid) ? W[(size_t)k * Ncols + col] : 0.f;
  out[t] = (_Float16)v;
}

// block-diagonal 64x64 from rel[4,16,16]: W[h*16+d][h*16+f] = rel[h][d][f]
__global__ void pack_bd_kernel(const float* __restrict__ rel, _Float16* __restrict__ out) {
  int t = blockIdx.x * blockDim.x + threadIdx.x;
  if (t >= 4096) return;
  int h = t & 15, lane = (t >> 4) & 31;
  int g = t >> 9;
  int kt = g & 1, ct = g >> 1;
  int k   = kt * 32 + ((lane >> 4) << 3) + (h & 7) + ((h >> 3) << 4);
  int col = ct * 16 + (lane & 15);
  int hk = k >> 4, hn = col >> 4;
  float v = (hk == hn) ? rel[((size_t)hk * 16 + (k & 15)) * 16 + (col & 15)] : 0.f;
  out[t] = (_Float16)v;
}

// ---------------------------------------------------------------------------
// WMMA GEMM: out[N,64] = act(A[N,64]) @ W[64,64] (+bias).
// 8 waves/block; each wave owns a full 16-row strip and iterates all 4 column
// tiles -> 8 v_wmma per A-fragment load. Compile-time SCATTER/GELU variants.
// ---------------------------------------------------------------------------
template<bool SCATTER, bool GELU>
__global__ __launch_bounds__(256) void gemm64_kernel(
    const float* __restrict__ A, const _Float16* __restrict__ Wp,
    const float* __restrict__ bias, float* __restrict__ out,
    const int* __restrict__ sidx, int nrows) {
  int wave = threadIdx.x >> 5;
  int lane = threadIdx.x & 31;
  int rowBase = (blockIdx.x * 8 + wave) * 16;
  if (rowBase >= nrows) return;
  int m    = lane & 15;
  int half = lane >> 4;
  int base = half * 8;        // A 16-bit layout: per-lane K = base+{0..7,16..23}
  int row  = rowBase + m;
  int rowc = row < nrows ? row : nrows - 1;
  const float* ap = A + (size_t)rowc * DM;
  v16h afrag[2];
#pragma unroll
  for (int kt = 0; kt < 2; ++kt) {
    const float* p0 = ap + kt * 32 + base;
#pragma unroll
    for (int i = 0; i < 8; ++i) {
      float x0 = p0[i];
      float x1 = p0[16 + i];
      if (GELU) { x0 = geluf_(x0); x1 = geluf_(x1); }
      afrag[kt][i]     = (_Float16)x0;
      afrag[kt][8 + i] = (_Float16)x1;
    }
  }
  v8f c[4];
#pragma unroll
  for (int ct = 0; ct < 4; ++ct) c[ct] = (v8f){0.f,0.f,0.f,0.f,0.f,0.f,0.f,0.f};
#pragma unroll
  for (int ct = 0; ct < 4; ++ct) {
#pragma unroll
    for (int kt = 0; kt < 2; ++kt) {
      v16h b = *(const v16h*)(Wp + (((size_t)(ct * 2 + kt)) * 32 + lane) * 16);
      c[ct] = __builtin_amdgcn_wmma_f32_16x16x32_f16(false, afrag[kt], false, b,
                                                     (short)0, c[ct], false, false);
    }
  }
  // epilogue: C layout VGPR v -> M = v (+8 for hi lanes)
  int idx[8];
  if (SCATTER) {
#pragma unroll
    for (int v = 0; v < 8; ++v) {
      int r = rowBase + v + half * 8;
      idx[v] = sidx[r < nrows ? r : nrows - 1];
    }
  }
  bool full = (rowBase + 16 <= nrows);
#pragma unroll
  for (int ct = 0; ct < 4; ++ct) {
    int col = ct * 16 + m;
    float bv = bias ? bias[col] : 0.f;
    if (full) {
#pragma unroll
      for (int v = 0; v < 8; ++v) {
        int r = rowBase + v + half * 8;
        float val = c[ct][v] + bv;
        if (SCATTER) atomicAdd(&out[(size_t)idx[v] * DM + col], val);
        else         out[(size_t)r * DM + col] = val;
      }
    } else {
#pragma unroll
      for (int v = 0; v < 8; ++v) {
        int r = rowBase + v + half * 8;
        if (r < nrows) {
          float val = c[ct][v] + bv;
          if (SCATTER) atomicAdd(&out[(size_t)idx[v] * DM + col], val);
          else         out[(size_t)r * DM + col] = val;
        }
      }
    }
  }
}

// ---------------------------------------------------------------------------
// Classifier GEMM: relu([h_p[src]|edge_attr|h_m[dst]] (K=140 pad 160) @ cls1[140,128] + b)
// 8 waves/block; each wave owns a 16-row strip, loops 5 k-tiles x 8 col tiles
// (40 v_wmma per strip); each gathered A fragment reused 8x. Output fp16.
// ---------------------------------------------------------------------------
static __device__ __forceinline__ float hcat_elem(const float* hps, const float* hmd,
                                                  const float* ea, int k) {
  if (k < 64)  return hps[k];
  if (k < 76)  return ea[k - 64];
  if (k < 140) return hmd[k - 76];
  return 0.f;
}

__global__ __launch_bounds__(256) void cls1_kernel(
    const float* __restrict__ hp, const float* __restrict__ hm,
    const float* __restrict__ eattr, const int* __restrict__ esrc,
    const int* __restrict__ edst, const _Float16* __restrict__ Wp,
    const float* __restrict__ bias, _Float16* __restrict__ outh, int E) {
  int wave = threadIdx.x >> 5;
  int lane = threadIdx.x & 31;
  int rowBase = (blockIdx.x * 8 + wave) * 16;
  if (rowBase >= E) return;
  int m = lane & 15, half = lane >> 4, base = half * 8;
  int e = rowBase + m;
  int ec = e < E ? e : E - 1;
  int s = esrc[ec], d = edst[ec];
  const float* hps = hp + (size_t)s * DM;
  const float* hmd = hm + (size_t)d * DM;
  const float* ea  = eattr + (size_t)ec * 12;
  v8f c[8];
#pragma unroll
  for (int ct = 0; ct < 8; ++ct) c[ct] = (v8f){0.f,0.f,0.f,0.f,0.f,0.f,0.f,0.f};
#pragma unroll
  for (int kt = 0; kt < 5; ++kt) {   // K = 160 (140 valid)
    v16h a;
#pragma unroll
    for (int i = 0; i < 8; ++i) {
      int k0 = kt * 32 + base + i;
      int k1 = k0 + 16;
      a[i]     = (_Float16)hcat_elem(hps, hmd, ea, k0);
      a[8 + i] = (_Float16)hcat_elem(hps, hmd, ea, k1);
    }
#pragma unroll
    for (int ct = 0; ct < 8; ++ct) {
      v16h b = *(const v16h*)(Wp + (((size_t)(ct * 5 + kt)) * 32 + lane) * 16);
      c[ct] = __builtin_amdgcn_wmma_f32_16x16x32_f16(false, a, false, b,
                                                     (short)0, c[ct], false, false);
    }
  }
  bool full = (rowBase + 16 <= E);
#pragma unroll
  for (int ct = 0; ct < 8; ++ct) {
    int col = ct * 16 + m;
    float bv = bias[col];
#pragma unroll
    for (int v = 0; v < 8; ++v) {
      int r = rowBase + v + half * 8;
      if (full || r < E) {
        float val = c[ct][v] + bv;
        val = val > 0.f ? val : 0.f;
        outh[(size_t)r * 128 + col] = (_Float16)val;
      }
    }
  }
}

// ---------------------------------------------------------------------------
// Node encoders
// ---------------------------------------------------------------------------
__global__ void encode_payer_kernel(const int* __restrict__ xp, const float* __restrict__ iss,
                                    const float* __restrict__ cty, const float* __restrict__ w,
                                    const float* __restrict__ b, float* __restrict__ h, int n) {
  int t = blockIdx.x * blockDim.x + threadIdx.x;
  if (t >= n * DM) return;
  int node = t >> 6, j = t & 63;
  int i0 = xp[node * 2], i1 = xp[node * 2 + 1];
  float acc = b[j];
#pragma unroll
  for (int k = 0; k < 16; ++k) acc += iss[(size_t)i0 * 16 + k] * w[k * DM + j];
#pragma unroll
  for (int k = 0; k < 8; ++k)  acc += cty[(size_t)i1 * 8 + k] * w[(16 + k) * DM + j];
  h[t] = acc;
}

__global__ void encode_merch_kernel(const int* __restrict__ xm, const float* __restrict__ mcc,
                                    const float* __restrict__ loc, const float* __restrict__ qri,
                                    const float* __restrict__ w, const float* __restrict__ b,
                                    float* __restrict__ h, int n) {
  int t = blockIdx.x * blockDim.x + threadIdx.x;
  if (t >= n * DM) return;
  int node = t >> 6, j = t & 63;
  int i0 = xm[node * 3], i1 = xm[node * 3 + 1], i2 = xm[node * 3 + 2];
  float acc = b[j];
#pragma unroll
  for (int k = 0; k < 16; ++k) acc += mcc[(size_t)i0 * 16 + k] * w[k * DM + j];
#pragma unroll
  for (int k = 0; k < 8; ++k)  acc += loc[(size_t)i1 * 8 + k] * w[(16 + k) * DM + j];
#pragma unroll
  for (int k = 0; k < 4; ++k)  acc += qri[(size_t)i2 * 4 + k] * w[(24 + k) * DM + j];
  h[t] = acc;
}

// ---------------------------------------------------------------------------
// Time encoder + segment means
// ---------------------------------------------------------------------------
__global__ void mm_init_kernel(float* mm) {
  if (threadIdx.x == 0 && blockIdx.x == 0) { mm[0] = 3.4e38f; mm[1] = -3.4e38f; }
}

__global__ void minmax_kernel(const float* __restrict__ t, float* mm, int n) {
  __shared__ float smn[256], smx[256];
  float lmn = 3.4e38f, lmx = -3.4e38f;
  for (int i = blockIdx.x * blockDim.x + threadIdx.x; i < n; i += gridDim.x * blockDim.x) {
    float v = t[i];
    lmn = fminf(lmn, v);
    lmx = fmaxf(lmx, v);
  }
  smn[threadIdx.x] = lmn; smx[threadIdx.x] = lmx;
  __syncthreads();
  for (int s = 128; s > 0; s >>= 1) {
    if (threadIdx.x < (unsigned)s) {
      smn[threadIdx.x] = fminf(smn[threadIdx.x], smn[threadIdx.x + s]);
      smx[threadIdx.x] = fmaxf(smx[threadIdx.x], smx[threadIdx.x + s]);
    }
    __syncthreads();
  }
  if (threadIdx.x == 0) { atomicMinF(&mm[0], smn[0]); atomicMaxF(&mm[1], smx[0]); }
}

__global__ void count_kernel(const int* __restrict__ src, const int* __restrict__ dst,
                             float* cp, float* cm, int E) {
  int e = blockIdx.x * blockDim.x + threadIdx.x;
  if (e >= E) return;
  atomicAdd(&cp[src[e]], 1.f);
  atomicAdd(&cm[dst[e]], 1.f);
}

__global__ void time_scatter_kernel(const float* __restrict__ ts, const int* __restrict__ src,
                                    const int* __restrict__ dst, const float* __restrict__ tw,
                                    const float* __restrict__ tb, const float* __restrict__ mm,
                                    float* accp, float* accm, int E) {
  int t = blockIdx.x * blockDim.x + threadIdx.x;
  if (t >= E * DM) return;
  int e = t >> 6, j = t & 63;
  float tmin = mm[0], tmax = mm[1], x = ts[e];
  float r = tmax - tmin;
  float tn = r > 0.f ? (x - tmin) / r : x;
  float c = __cosf(tn * tw[j] + tb[j]);
  atomicAdd(&accp[(size_t)src[e] * DM + j], c);
  atomicAdd(&accm[(size_t)dst[e] * DM + j], c);
}

__global__ void divcnt_kernel(float* __restrict__ dst, const float* __restrict__ acc,
                              const float* __restrict__ cnt, int n, int add) {
  int t = blockIdx.x * blockDim.x + threadIdx.x;
  if (t >= n * DM) return;
  float c = cnt[t >> 6];
  c = c > 1.f ? c : 1.f;
  float v = acc[t] / c;
  if (add) dst[t] += v; else dst[t] = v;
}

__global__ void add_kernel(float* __restrict__ d, const float* __restrict__ s, int n) {
  int t = blockIdx.x * blockDim.x + threadIdx.x;
  if (t < n) d[t] += s[t];
}

__global__ void fill_kernel(unsigned int* p, unsigned int v, int n) {
  int t = blockIdx.x * blockDim.x + threadIdx.x;
  if (t < n) p[t] = v;
}

// edge_feat_proj: [E,12] @ [12,64] + b
__global__ void edge_proj_kernel(const float* __restrict__ ea, const float* __restrict__ w,
                                 const float* __restrict__ b, float* __restrict__ out, int E) {
  int t = blockIdx.x * blockDim.x + threadIdx.x;
  if (t >= E * DM) return;
  int e = t >> 6, j = t & 63;
  const float* x = ea + (size_t)e * 12;
  float acc = b[j];
#pragma unroll
  for (int k = 0; k < 12; ++k) acc += x[k] * w[k * DM + j];
  out[t] = acc;
}

// ---------------------------------------------------------------------------
// HGT edge attention (softmax over destination segments)
// ---------------------------------------------------------------------------
__global__ void score_kernel(const float* __restrict__ q, const float* __restrict__ kr,
                             const float* __restrict__ prel, const int* __restrict__ srci,
                             const int* __restrict__ dsti, float* __restrict__ score,
                             float* smax, int E) {
  int t = blockIdx.x * blockDim.x + threadIdx.x;
  if (t >= E * 4) return;
  int e = t >> 2, h = t & 3;
  int s = srci[e], d = dsti[e];
  const float* qr = q + (size_t)d * DM + h * 16;
  const float* kp = kr + (size_t)s * DM + h * 16;
  float acc = 0.f;
#pragma unroll
  for (int i = 0; i < 16; ++i) acc += qr[i] * kp[i];
  acc *= prel[h] * 0.25f;   // 1/sqrt(HEAD_D)
  score[t] = acc;
  atomicMaxF(&smax[(size_t)d * 4 + h], acc);
}

__global__ void exp_kernel(float* __restrict__ score, const int* __restrict__ dsti,
                           const float* __restrict__ smax, float* den, int E) {
  int t = blockIdx.x * blockDim.x + threadIdx.x;
  if (t >= E * 4) return;
  int e = t >> 2, h = t & 3;
  int d = dsti[e];
  float ex = __expf(score[t] - smax[(size_t)d * 4 + h]);
  score[t] = ex;
  atomicAdd(&den[(size_t)d * 4 + h], ex);
}

__global__ void scatter_v_kernel(const float* __restrict__ vr, const float* __restrict__ ex,
                                 const float* __restrict__ den, const int* __restrict__ srci,
                                 const int* __restrict__ dsti, float* agg, int E) {
  int t = blockIdx.x * blockDim.x + threadIdx.x;
  if (t >= E * DM) return;
  int e = t >> 6, j = t & 63, h = j >> 4;
  int s = srci[e], d = dsti[e];
  float alpha = ex[(size_t)e * 4 + h] / (den[(size_t)d * 4 + h] + 1e-16f);
  atomicAdd(&agg[(size_t)d * DM + j], vr[(size_t)s * DM + j] * alpha);
}

// ---------------------------------------------------------------------------
// skip-gate + residual + LayerNorm: one wave32 per node (lane handles j, j+32)
// ---------------------------------------------------------------------------
__global__ void skip_ln_kernel(float* __restrict__ h, const float* __restrict__ outb,
                               const float* __restrict__ res, const float* __restrict__ skip,
                               const float* __restrict__ g, const float* __restrict__ b, int n) {
  int node = blockIdx.x * 8 + (threadIdx.x >> 5);
  int lane = threadIdx.x & 31;
  if (node >= n) return;
  float beta = sigmoidf_(skip[0]);
  size_t o = (size_t)node * DM;
  float x0 = beta * outb[o + lane]      + (1.f - beta) * h[o + lane]      + res[o + lane];
  float x1 = beta * outb[o + lane + 32] + (1.f - beta) * h[o + lane + 32] + res[o + lane + 32];
  float ss = x0 + x1;
  for (int off = 16; off > 0; off >>= 1) ss += __shfl_xor(ss, off, 32);
  float mu = ss * (1.f / 64.f);
  float d0 = x0 - mu, d1 = x1 - mu;
  float vs = d0 * d0 + d1 * d1;
  for (int off = 16; off > 0; off >>= 1) vs += __shfl_xor(vs, off, 32);
  float inv = rsqrtf(vs * (1.f / 64.f) + 1e-5f);
  h[o + lane]      = d0 * inv * g[lane]      + b[lane];
  h[o + lane + 32] = d1 * inv * g[lane + 32] + b[lane + 32];
}

// ---------------------------------------------------------------------------
// Final: gnn logit (dot with cls2) + anomaly MLP + gate
// ---------------------------------------------------------------------------
__global__ void final_kernel(const _Float16* __restrict__ c1h, const float* __restrict__ c2w,
                             const float* __restrict__ c2b, const float* __restrict__ eattr,
                             const float* __restrict__ a1w, const float* __restrict__ a1b,
                             const float* __restrict__ a2w, const float* __restrict__ a2b,
                             const float* __restrict__ a3w, const float* __restrict__ a3b,
                             const float* __restrict__ gate, float* __restrict__ out, int E) {
  int e = blockIdx.x * blockDim.x + threadIdx.x;
  if (e >= E) return;
  float g = sigmoidf_(gate[0]);
  const _Float16* r = c1h + (size_t)e * 128;
  float gnn = c2b[0];
#pragma unroll 8
  for (int i = 0; i < 128; ++i) gnn += (float)r[i] * c2w[i];
  // anomaly MLP 12 -> 64 -> 32 -> 1
  const float* x = eattr + (size_t)e * 12;
  float xr[12];
#pragma unroll
  for (int k = 0; k < 12; ++k) xr[k] = x[k];
  float a1[64];
#pragma unroll
  for (int j = 0; j < 64; ++j) {
    float s = a1b[j];
#pragma unroll
    for (int k = 0; k < 12; ++k) s += xr[k] * a1w[k * 64 + j];
    a1[j] = s > 0.f ? s : 0.f;
  }
  float a3 = a3b[0];
  for (int i = 0; i < 32; ++i) {
    float s2 = a2b[i];
#pragma unroll
    for (int j = 0; j < 64; ++j) s2 += a1[j] * a2w[j * 32 + i];
    s2 = s2 > 0.f ? s2 : 0.f;
    a3 += s2 * a3w[i];
  }
  out[e] = g * gnn + (1.f - g) * a3;
}

// ---------------------------------------------------------------------------
// Host orchestration
// ---------------------------------------------------------------------------
static inline int cdiv_(int a, int b) { return (a + b - 1) / b; }

extern "C" void kernel_launch(void* const* d_in, const int* in_sizes, int n_in,
                              void* d_out, int out_size, void* d_ws, size_t ws_size,
                              hipStream_t stream) {
  (void)in_sizes; (void)n_in; (void)out_size; (void)ws_size;
  const int NP = NPAYER, NM = NMERCH, NE = NEDGE;

  const int*   x_payer = (const int*)d_in[0];
  const int*   x_merch = (const int*)d_in[1];
  const int*   esrc    = (const int*)d_in[2];
  const int*   edst    = (const int*)d_in[3];
  const float* eattr   = (const float*)d_in[4];
  const float* ts      = (const float*)d_in[5];
  // params: JAX tree-flatten (sorted dict keys), leaves after the 6 arrays.
  const float* const* P = (const float* const*)(d_in + 6);
  enum {
    AN1_B = 0, AN1_W, AN2_B, AN2_W, AN3_B, AN3_W,
    CLS1_B, CLS1_W, CLS2_B, CLS2_W,
    COUNTRY, EPROJ_B, EPROJ_W, E2D_B, E2D_W, E2S_B, E2S_W, GATE, ISSUER,
    LBASE  // = 19 ; 3 layers x 28 leaves follow
  };
  const int LOC = LBASE + 84, MCC = LOC + 1, MERCH_B = MCC + 1, MERCH_W = MERCH_B + 1,
            PAYER_B = MERCH_W + 1, PAYER_W = PAYER_B + 1, QRIS = PAYER_W + 1,
            TIME_B = QRIS + 1, TIME_W = TIME_B + 1;
  // per-layer (sorted): 0 a_m.b 1 a_m.w 2 a_p.b 3 a_p.w 4 arel_mp 5 arel_pm
  // 6 k_m.b 7 k_m.w 8 k_p.b 9 k_p.w 10 ln_m_b 11 ln_m_g 12 ln_p_b 13 ln_p_g
  // 14 mrel_mp 15 mrel_pm 16 prel_mp 17 prel_pm 18 q_m.b 19 q_m.w 20 q_p.b
  // 21 q_p.w 22 skip_m 23 skip_p 24 v_m.b 25 v_m.w 26 v_p.b 27 v_p.w

  // ---- workspace layout --------------------------------------------------
  char* wsb = (char*)d_ws;
  size_t off = 0;
  auto alloc = [&](size_t bytes) -> void* {
    void* p = wsb + off;
    off += (bytes + 255) & ~(size_t)255;
    return p;
  };
  const size_t FP = (size_t)NP * DM * sizeof(float);
  const size_t FM = (size_t)NM * DM * sizeof(float);
  float* h_p    = (float*)alloc(FP);
  float* h_m    = (float*)alloc(FM);
  float* res_p  = (float*)alloc(FP);
  float* res_m  = (float*)alloc(FM);
  float* kp     = (float*)alloc(FP);
  float* qp     = (float*)alloc(FP);
  float* vp     = (float*)alloc(FP);
  float* krp    = (float*)alloc(FP);
  float* vrp    = (float*)alloc(FP);
  float* aggp   = (float*)alloc(FP);
  float* km     = (float*)alloc(FM);
  float* qm     = (float*)alloc(FM);
  float* vm     = (float*)alloc(FM);
  float* krm    = (float*)alloc(FM);
  float* vrm    = (float*)alloc(FM);
  float* aggm   = (float*)alloc(FM);
  float* mean_s = (float*)alloc(FP);
  float* mean_d = (float*)alloc(FM);
  float* acc_p  = (float*)alloc(FP);
  float* acc_m  = (float*)alloc(FM);
  float* cnt_p  = (float*)alloc((size_t)NP * sizeof(float));
  float* cnt_m  = (float*)alloc((size_t)NM * sizeof(float));
  float* smax_p = (float*)alloc((size_t)NP * 4 * sizeof(float));
  float* den_p  = (float*)alloc((size_t)NP * 4 * sizeof(float));
  float* smax_m = (float*)alloc((size_t)NM * 4 * sizeof(float));
  float* den_m  = (float*)alloc((size_t)NM * 4 * sizeof(float));
  float* mm     = (float*)alloc(256);
  float* score  = (float*)alloc((size_t)NE * 4 * sizeof(float));
  float* efp    = (float*)alloc((size_t)NE * DM * sizeof(float)); // later reused as fp16 [NE,128]
  // packed fp16 weights: 2 + 36 (64x64) + 1 (160x128)
  _Float16* pwb = (_Float16*)alloc((size_t)(38 * 4096 + 20480) * sizeof(_Float16));
  _Float16* PW_ESRC = pwb;
  _Float16* PW_EDST = pwb + 4096;
  auto PWL = [&](int l, int midx) { return pwb + 8192 + (size_t)(l * 12 + midx) * 4096; };
  // midx: 0 KP 1 QP 2 VP 3 AP 4 KM 5 QM 6 VM 7 AM 8 BDA_PM 9 BDM_PM 10 BDA_MP 11 BDM_MP
  _Float16* PW_CLS1 = pwb + 8192 + (size_t)36 * 4096;

  // ---- pack weights ------------------------------------------------------
  pack_b_kernel<<<16, 256, 0, stream>>>(P[E2S_W], PW_ESRC, 64, 2, 4, 64);
  pack_b_kernel<<<16, 256, 0, stream>>>(P[E2D_W], PW_EDST, 64, 2, 4, 64);
  for (int l = 0; l < 3; ++l) {
    const float* const* Lp = P + LBASE + 28 * l;
    pack_b_kernel<<<16, 256, 0, stream>>>(Lp[9],  PWL(l, 0), 64, 2, 4, 64);  // k_p.w
    pack_b_kernel<<<16, 256, 0, stream>>>(Lp[21], PWL(l, 1), 64, 2, 4, 64);  // q_p.w
    pack_b_kernel<<<16, 256, 0, stream>>>(Lp[27], PWL(l, 2), 64, 2, 4, 64);  // v_p.w
    pack_b_kernel<<<16, 256, 0, stream>>>(Lp[3],  PWL(l, 3), 64, 2, 4, 64);  // a_p.w
    pack_b_kernel<<<16, 256, 0, stream>>>(Lp[7],  PWL(l, 4), 64, 2, 4, 64);  // k_m.w
    pack_b_kernel<<<16, 256, 0, stream>>>(Lp[19], PWL(l, 5), 64, 2, 4, 64);  // q_m.w
    pack_b_kernel<<<16, 256, 0, stream>>>(Lp[25], PWL(l, 6), 64, 2, 4, 64);  // v_m.w
    pack_b_kernel<<<16, 256, 0, stream>>>(Lp[1],  PWL(l, 7), 64, 2, 4, 64);  // a_m.w
    pack_bd_kernel<<<16, 256, 0, stream>>>(Lp[5],  PWL(l, 8));               // arel_pm
    pack_bd_kernel<<<16, 256, 0, stream>>>(Lp[15], PWL(l, 9));               // mrel_pm
    pack_bd_kernel<<<16, 256, 0, stream>>>(Lp[4],  PWL(l, 10));              // arel_mp
    pack_bd_kernel<<<16, 256, 0, stream>>>(Lp[14], PWL(l, 11));              // mrel_mp
  }
  pack_b_kernel<<<80, 256, 0, stream>>>(P[CLS1_W], PW_CLS1, 140, 5, 8, 128);

  const int gNP = cdiv_(NP * DM, 256), gNM = cdiv_(NM * DM, 256);
  const int gE64 = cdiv_(NE * DM, 256), gE4 = cdiv_(NE * 4, 256), gE = cdiv_(NE, 256);

  // ---- node encoders -----------------------------------------------------
  encode_payer_kernel<<<gNP, 256, 0, stream>>>(x_payer, P[ISSUER], P[COUNTRY],
                                               P[PAYER_W], P[PAYER_B], h_p, NP);
  encode_merch_kernel<<<gNM, 256, 0, stream>>>(x_merch, P[MCC], P[LOC], P[QRIS],
                                               P[MERCH_W], P[MERCH_B], h_m, NM);

  // ---- time encoder + segment means --------------------------------------
  mm_init_kernel<<<1, 32, 0, stream>>>(mm);
  minmax_kernel<<<512, 256, 0, stream>>>(ts, mm, NE);
  hipMemsetAsync(cnt_p, 0, (size_t)NP * sizeof(float), stream);
  hipMemsetAsync(cnt_m, 0, (size_t)NM * sizeof(float), stream);
  hipMemsetAsync(acc_p, 0, FP, stream);
  hipMemsetAsync(acc_m, 0, FM, stream);
  count_kernel<<<gE, 256, 0, stream>>>(esrc, edst, cnt_p, cnt_m, NE);
  time_scatter_kernel<<<gE64, 256, 0, stream>>>(ts, esrc, edst, P[TIME_W], P[TIME_B],
                                                mm, acc_p, acc_m, NE);
  divcnt_kernel<<<gNP, 256, 0, stream>>>(h_p, acc_p, cnt_p, NP, 1);
  divcnt_kernel<<<gNM, 256, 0, stream>>>(h_m, acc_m, cnt_m, NM, 1);

  // ---- edge_feat_proj + layer-invariant edge_to_src/dst means (WMMA) ------
  edge_proj_kernel<<<gE64, 256, 0, stream>>>(eattr, P[EPROJ_W], P[EPROJ_B], efp, NE);
  hipMemsetAsync(acc_p, 0, FP, stream);
  gemm64_kernel<true, false><<<cdiv_(NE, 128), 256, 0, stream>>>(efp, PW_ESRC, P[E2S_B], acc_p, esrc, NE);
  divcnt_kernel<<<gNP, 256, 0, stream>>>(mean_s, acc_p, cnt_p, NP, 0);
  hipMemsetAsync(acc_m, 0, FM, stream);
  gemm64_kernel<true, false><<<cdiv_(NE, 128), 256, 0, stream>>>(efp, PW_EDST, P[E2D_B], acc_m, edst, NE);
  divcnt_kernel<<<gNM, 256, 0, stream>>>(mean_d, acc_m, cnt_m, NM, 0);

  // ---- HGT layers ---------------------------------------------------------
  const int gP128 = cdiv_(NP, 128), gM128 = cdiv_(NM, 128);
  for (int l = 0; l < 3; ++l) {
    const float* const* Lp = P + LBASE + 28 * l;
    hipMemcpyAsync(res_p, h_p, FP, hipMemcpyDeviceToDevice, stream);
    hipMemcpyAsync(res_m, h_m, FM, hipMemcpyDeviceToDevice, stream);
    add_kernel<<<gNP, 256, 0, stream>>>(h_p, mean_s, NP * DM);
    add_kernel<<<gNM, 256, 0, stream>>>(h_m, mean_d, NM * DM);
    // K/Q/V linears (WMMA)
    gemm64_kernel<false, false><<<gP128, 256, 0, stream>>>(h_p, PWL(l, 0), Lp[8],  kp, nullptr, NP);
    gemm64_kernel<false, false><<<gP128, 256, 0, stream>>>(h_p, PWL(l, 1), Lp[20], qp, nullptr, NP);
    gemm64_kernel<false, false><<<gP128, 256, 0, stream>>>(h_p, PWL(l, 2), Lp[26], vp, nullptr, NP);
    gemm64_kernel<false, false><<<gM128, 256, 0, stream>>>(h_m, PWL(l, 4), Lp[6],  km, nullptr, NM);
    gemm64_kernel<false, false><<<gM128, 256, 0, stream>>>(h_m, PWL(l, 5), Lp[18], qm, nullptr, NM);
    gemm64_kernel<false, false><<<gM128, 256, 0, stream>>>(h_m, PWL(l, 6), Lp[24], vm, nullptr, NM);
    // relation transforms as node-level block-diagonal GEMMs (WMMA)
    gemm64_kernel<false, false><<<gP128, 256, 0, stream>>>(kp, PWL(l, 8),  nullptr, krp, nullptr, NP);
    gemm64_kernel<false, false><<<gP128, 256, 0, stream>>>(vp, PWL(l, 9),  nullptr, vrp, nullptr, NP);
    gemm64_kernel<false, false><<<gM128, 256, 0, stream>>>(km, PWL(l, 10), nullptr, krm, nullptr, NM);
    gemm64_kernel<false, false><<<gM128, 256, 0, stream>>>(vm, PWL(l, 11), nullptr, vrm, nullptr, NM);
    // payer -> merchant (segment softmax over edge_dst)
    fill_kernel<<<cdiv_(NM * 4, 256), 256, 0, stream>>>((unsigned int*)smax_m, 0xFF800000u, NM * 4);
    hipMemsetAsync(den_m, 0, (size_t)NM * 4 * sizeof(float), stream);
    hipMemsetAsync(aggm, 0, FM, stream);
    score_kernel<<<gE4, 256, 0, stream>>>(qm, krp, Lp[17], esrc, edst, score, smax_m, NE);
    exp_kernel<<<gE4, 256, 0, stream>>>(score, edst, smax_m, den_m, NE);
    scatter_v_kernel<<<gE64, 256, 0, stream>>>(vrp, score, den_m, esrc, edst, aggm, NE);
    // merchant -> payer (segment softmax over edge_src)
    fill_kernel<<<cdiv_(NP * 4, 256), 256, 0, stream>>>((unsigned int*)smax_p, 0xFF800000u, NP * 4);
    hipMemsetAsync(den_p, 0, (size_t)NP * 4 * sizeof(float), stream);
    hipMemsetAsync(aggp, 0, FP, stream);
    score_kernel<<<gE4, 256, 0, stream>>>(qp, krm, Lp[16], edst, esrc, score, smax_p, NE);
    exp_kernel<<<gE4, 256, 0, stream>>>(score, esrc, smax_p, den_p, NE);
    scatter_v_kernel<<<gE64, 256, 0, stream>>>(vrm, score, den_p, edst, esrc, aggp, NE);
    // out projection with fused exact-GELU on A, then skip-gate + residual + LN
    gemm64_kernel<false, true><<<gP128, 256, 0, stream>>>(aggp, PWL(l, 3), Lp[2], kp, nullptr, NP);
    gemm64_kernel<false, true><<<gM128, 256, 0, stream>>>(aggm, PWL(l, 7), Lp[0], km, nullptr, NM);
    skip_ln_kernel<<<cdiv_(NP, 8), 256, 0, stream>>>(h_p, kp, res_p, Lp[23], Lp[13], Lp[12], NP);
    skip_ln_kernel<<<cdiv_(NM, 8), 256, 0, stream>>>(h_m, km, res_m, Lp[22], Lp[11], Lp[10], NM);
  }

  // ---- classifier (WMMA, gathered hcat, fp16 intermediate in efp buffer) ---
  _Float16* c1h = (_Float16*)efp;  // NE*128 halves == NE*64 floats
  cls1_kernel<<<cdiv_(NE, 128), 256, 0, stream>>>(h_p, h_m, eattr, esrc, edst,
                                                  PW_CLS1, P[CLS1_B], c1h, NE);
  final_kernel<<<gE, 256, 0, stream>>>(c1h, P[CLS2_W], P[CLS2_B], eattr,
                                       P[AN1_W], P[AN1_B], P[AN2_W], P[AN2_B],
                                       P[AN3_W], P[AN3_B], P[GATE], (float*)d_out, NE);
}
